// ManualLSTM_74466142978188
// MI455X (gfx1250) — matrix operand order repaired and checked by reading, not compiled, not run on hardware
//
#include <hip/hip_runtime.h>
#include <hip/hip_bf16.h>

// ---------------------------------------------------------------------------
// Problem constants (from the reference)
// ---------------------------------------------------------------------------
#define T_STEPS 512
#define BATCH   32
#define IN_DIM  1024
#define H_DIM   1024
#define OUT_DIM 512
#define K_TOT   (IN_DIM + H_DIM)   // 2048
#define FOUR_H  (4 * H_DIM)        // 4096
#define KCHUNK  (K_TOT / 4)        // 512: K-range owned by one wave-chunk
#define NKITER  (KCHUNK / 32)      // 16 WMMA k-steps per wave

typedef __attribute__((ext_vector_type(16))) __bf16          v16bf;
typedef __attribute__((ext_vector_type(8)))  float           v8f;
typedef __attribute__((ext_vector_type(8)))  unsigned short  su8;
typedef __attribute__((ext_vector_type(16))) unsigned short  su16;

// fp32 -> bf16, round-to-nearest-even
__device__ __forceinline__ unsigned short f2bf(float f) {
    unsigned u = __builtin_bit_cast(unsigned, f);
    unsigned r = u + 0x7FFFu + ((u >> 16) & 1u);
    return (unsigned short)(r >> 16);
}

__device__ __forceinline__ float fast_sigmoid(float x) {
    return 1.0f / (1.0f + __expf(-x));
}
__device__ __forceinline__ float fast_tanh(float x) {
    x = fminf(fmaxf(x, -15.0f), 15.0f);
    float e = __expf(-2.0f * x);
    return (1.0f - e) / (1.0f + e);
}

// ---------------------------------------------------------------------------
// WMMA operand loaders (bf16, 16x16x32).  ISA 7.12.2 layouts:
//   A (16x32, MxK): lane l -> M = l&15, khalf = l>>4;
//     slots 0..7  = K (khalf*8 .. +7),  slots 8..15 = K (16+khalf*8 .. +7)
//   B (32x16, KxN): lane l -> N = l&15, khalf = l>>4;
//     slots 0..7  = K (khalf*8 .. +7),  slots 8..15 = K (16+khalf*8 .. +7)
//     (same K order as A) -> row-major W rows load as B operand of A @ W^T.
// ---------------------------------------------------------------------------
__device__ __forceinline__ v16bf load_a_bf16(const unsigned short* __restrict__ base,
                                             int ldk, int m, int k, int kh) {
    const su8* p0 = (const su8*)(base + (size_t)m * ldk + k + kh * 8);
    const su8* p1 = (const su8*)(base + (size_t)m * ldk + k + 16 + kh * 8);
    su8 lo = *p0, hi = *p1;
    su16 u;
#pragma unroll
    for (int i = 0; i < 8; ++i) { u[i] = lo[i]; u[i + 8] = hi[i]; }
    return __builtin_bit_cast(v16bf, u);
}

__device__ __forceinline__ v16bf load_b_bf16(const unsigned short* __restrict__ wrow,
                                             int k, int kh) {
    const su8* p0 = (const su8*)(wrow + k + kh * 8);
    const su8* p1 = (const su8*)(wrow + k + 16 + kh * 8);
    su8 lo = *p0, hi = *p1;
    su16 u;
#pragma unroll
    for (int i = 0; i < 8; ++i) { u[i] = lo[i]; u[i + 8] = hi[i]; }
    return __builtin_bit_cast(v16bf, u);
}

__device__ __forceinline__ v8f wmma_bf16(v16bf a, v16bf b, v8f c) {
    return __builtin_amdgcn_wmma_f32_16x16x32_bf16(
        /*neg_a=*/false, a, /*neg_b=*/false, b,
        /*c_mod=*/(short)0, c, /*reuse_a=*/false, /*reuse_b=*/false);
}

// ---------------------------------------------------------------------------
// Kernel: fp32 -> bf16 conversion (grid-stride)
// ---------------------------------------------------------------------------
__global__ void cvt_f32_to_bf16(const float* __restrict__ src,
                                unsigned short* __restrict__ dst, size_t n) {
    size_t i = (size_t)blockIdx.x * blockDim.x + threadIdx.x;
    size_t stride = (size_t)gridDim.x * blockDim.x;
    for (; i < n; i += stride) dst[i] = f2bf(src[i]);
}

// ---------------------------------------------------------------------------
// Kernel: initialize LSTM state from h0 / c0
// ---------------------------------------------------------------------------
__global__ void init_state(const float* __restrict__ h0, const float* __restrict__ c0,
                           unsigned short* __restrict__ h_bf0,
                           float* __restrict__ h_f32, float* __restrict__ c_f32) {
    int i = blockIdx.x * blockDim.x + threadIdx.x;
    if (i < BATCH * H_DIM) {
        float h = h0[i];
        h_bf0[i] = f2bf(h);
        h_f32[i] = h;
        c_f32[i] = c0[i];
    }
}

// ---------------------------------------------------------------------------
// Kernel: one LSTM timestep.
//   grid  = H/16 = 64 blocks (each owns a 16-wide slice of the H columns)
//   block = 512 threads = 16 waves; wave w -> gate (w&3), K-chunk (w>>2).
//   Each wave runs K=512 (16 WMMA k-steps) with a DISTANCE-2 software
//   pipeline (operand sets for k, k+1, k+2 in flight) so no WMMA has to
//   wait for loadcnt==0.  The 4 K-chunk partials per gate are summed in
//   LDS in a fixed order (deterministic).
//   K-chunks 0,1 read h (cols 0..1023); chunks 2,3 read x_t (cols 1024..2047).
// ---------------------------------------------------------------------------
__global__ __launch_bounds__(512) void lstm_step(
    const unsigned short* __restrict__ Wg_bf,   // [4096][2048] bf16 row-major
    const float* __restrict__ bg,               // [4096]
    const unsigned short* __restrict__ x_bf_t,  // [32][1024] bf16 (this t)
    const unsigned short* __restrict__ h_in_bf, // [32][1024] bf16
    unsigned short* __restrict__ h_out_bf,      // [32][1024] bf16
    float* __restrict__ h_f32,                  // [32][1024]
    float* __restrict__ c_f32,                  // [32][1024]
    unsigned short* __restrict__ out_bf_t)      // [32][1024] bf16 (outs row t)
{
    __shared__ float pre[4][4][BATCH][16];      // [gate][chunk][m][n] = 32 KB

    const int wave  = threadIdx.x >> 5;
    const int lane  = threadIdx.x & 31;
    const int gate  = wave & 3;                 // 0=f 1=i 2=g 3=o
    const int chunk = wave >> 2;                // 0..3 -> K range
    const int nl    = lane & 15;
    const int kh    = lane >> 4;
    const int nTile = blockIdx.x;               // 0..63

    const unsigned short* brow =
        Wg_bf + ((size_t)gate * H_DIM + (size_t)nTile * 16 + nl) * K_TOT;

    // K source for this chunk (concat boundary falls between chunks 1 and 2)
    const int kbase = chunk * KCHUNK;
    const unsigned short* asrc = (chunk < 2) ? h_in_bf : x_bf_t;
    const int abase = (chunk < 2) ? kbase : (kbase - H_DIM);

    v8f acc0 = {};   // batch rows 0..15
    v8f acc1 = {};   // batch rows 16..31

    // ---- distance-2 software pipeline ----
    v16bf a0c = load_a_bf16(asrc, H_DIM, nl,      abase, kh);
    v16bf a1c = load_a_bf16(asrc, H_DIM, nl + 16, abase, kh);
    v16bf bc  = load_b_bf16(brow, kbase, kh);
    v16bf a0n = load_a_bf16(asrc, H_DIM, nl,      abase + 32, kh);
    v16bf a1n = load_a_bf16(asrc, H_DIM, nl + 16, abase + 32, kh);
    v16bf bn  = load_b_bf16(brow, kbase + 32, kh);

#pragma unroll 7
    for (int kc = 0; kc < NKITER - 2; ++kc) {
        const int kf = (kc + 2) * 32;
        v16bf a0f = load_a_bf16(asrc, H_DIM, nl,      abase + kf, kh);
        v16bf a1f = load_a_bf16(asrc, H_DIM, nl + 16, abase + kf, kh);
        v16bf bf_ = load_b_bf16(brow, kbase + kf, kh);
        acc0 = wmma_bf16(a0c, bc, acc0);
        acc1 = wmma_bf16(a1c, bc, acc1);
        a0c = a0n; a1c = a1n; bc = bn;
        a0n = a0f; a1n = a1f; bn = bf_;
    }
    acc0 = wmma_bf16(a0c, bc, acc0);
    acc1 = wmma_bf16(a1c, bc, acc1);
    acc0 = wmma_bf16(a0n, bn, acc0);
    acc1 = wmma_bf16(a1n, bn, acc1);

    // C/D layout: VGPR r -> M = r + 8*kh (per lane half), N = nl
#pragma unroll
    for (int r = 0; r < 8; ++r) {
        pre[gate][chunk][r + 8 * kh][nl]      = acc0[r];
        pre[gate][chunk][16 + r + 8 * kh][nl] = acc1[r];
    }
    __syncthreads();

    // Fused gate nonlinearities + state update: 512 cells, one per thread.
    // Chunk partials summed in fixed order -> deterministic.
    {
        const int idx = threadIdx.x;
        const int m   = idx >> 4;
        const int n   = idx & 15;
        const int ng  = nTile * 16 + n;         // global column in H

        float fp = ((pre[0][0][m][n] + pre[0][1][m][n]) +
                    (pre[0][2][m][n] + pre[0][3][m][n])) + bg[0 * H_DIM + ng];
        float ip = ((pre[1][0][m][n] + pre[1][1][m][n]) +
                    (pre[1][2][m][n] + pre[1][3][m][n])) + bg[1 * H_DIM + ng];
        float gp = ((pre[2][0][m][n] + pre[2][1][m][n]) +
                    (pre[2][2][m][n] + pre[2][3][m][n])) + bg[2 * H_DIM + ng];
        float op = ((pre[3][0][m][n] + pre[3][1][m][n]) +
                    (pre[3][2][m][n] + pre[3][3][m][n])) + bg[3 * H_DIM + ng];

        float ft = fast_sigmoid(fp);
        float it = fast_sigmoid(ip);
        float gt = fast_tanh(gp);
        float ot = fast_sigmoid(op);

        const size_t off = (size_t)m * H_DIM + ng;
        float cn = ft * c_f32[off] + it * gt;
        float hn = ot * fast_tanh(cn);

        c_f32[off] = cn;
        h_f32[off] = hn;
        unsigned short hb = f2bf(hn);
        h_out_bf[off] = hb;
        out_bf_t[off] = hb;
    }
}

// ---------------------------------------------------------------------------
// Kernel: output projection + log_softmax, fused.
//   grid  = (T*B)/16 = 1024 blocks (16 rows each)
//   block = 256 threads = 8 waves; wave w owns columns [w*64, w*64+64)
//   Distance-1 pipeline; latency hidden by the 1024 independent blocks.
// ---------------------------------------------------------------------------
__global__ __launch_bounds__(256) void proj_logsoftmax(
    const unsigned short* __restrict__ outs_bf, // [T*B][H] bf16
    const unsigned short* __restrict__ Wo_bf,   // [OUT][H] bf16
    const float* __restrict__ bo,               // [OUT]
    float* __restrict__ logp)                   // [T*B][OUT]
{
    __shared__ float lg[16][OUT_DIM];
    __shared__ float rowMax[16], rowLse[16];

    const int wave = threadIdx.x >> 5;
    const int lane = threadIdx.x & 31;
    const int nl   = lane & 15;
    const int kh   = lane >> 4;
    const int mt   = blockIdx.x;           // row tile (16 rows)

    const unsigned short* arow = outs_bf + (size_t)mt * 16 * H_DIM;
    const unsigned short* brow[4];
#pragma unroll
    for (int i = 0; i < 4; ++i)
        brow[i] = Wo_bf + (size_t)(wave * 64 + i * 16 + nl) * H_DIM;

    v8f acc[4] = {};

    v16bf a = load_a_bf16(arow, H_DIM, nl, 0, kh);
    v16bf b[4];
#pragma unroll
    for (int i = 0; i < 4; ++i) b[i] = load_b_bf16(brow[i], 0, kh);

#pragma unroll 2
    for (int kb = 0; kb < H_DIM / 32 - 1; ++kb) {
        const int kn = (kb + 1) * 32;
        v16bf na = load_a_bf16(arow, H_DIM, nl, kn, kh);
        v16bf nb[4];
#pragma unroll
        for (int i = 0; i < 4; ++i) nb[i] = load_b_bf16(brow[i], kn, kh);
#pragma unroll
        for (int i = 0; i < 4; ++i) acc[i] = wmma_bf16(a, b[i], acc[i]);
        a = na;
#pragma unroll
        for (int i = 0; i < 4; ++i) b[i] = nb[i];
    }
#pragma unroll
    for (int i = 0; i < 4; ++i) acc[i] = wmma_bf16(a, b[i], acc[i]);

#pragma unroll
    for (int i = 0; i < 4; ++i) {
        const int ncol = wave * 64 + i * 16;
#pragma unroll
        for (int r = 0; r < 8; ++r)
            lg[r + 8 * kh][ncol + nl] = acc[i][r] + bo[ncol + nl];
    }
    __syncthreads();

    // Row-wise logsumexp: wave w reduces rows 2w and 2w+1 via lane shuffles
    for (int rr = 0; rr < 2; ++rr) {
        const int m = wave * 2 + rr;
        float mx = -3.0e38f;
        for (int c = lane; c < OUT_DIM; c += 32) mx = fmaxf(mx, lg[m][c]);
#pragma unroll
        for (int off = 16; off; off >>= 1) mx = fmaxf(mx, __shfl_xor(mx, off, 32));
        float s = 0.0f;
        for (int c = lane; c < OUT_DIM; c += 32) s += __expf(lg[m][c] - mx);
#pragma unroll
        for (int off = 16; off; off >>= 1) s += __shfl_xor(s, off, 32);
        if (lane == 0) { rowMax[m] = mx; rowLse[m] = __logf(s); }
    }
    __syncthreads();

    const size_t base = (size_t)mt * 16 * OUT_DIM;
    for (int idx = threadIdx.x; idx < 16 * OUT_DIM; idx += 256) {
        const int m = idx >> 9;          // OUT_DIM == 512
        logp[base + idx] = lg[m][idx & (OUT_DIM - 1)] - rowMax[m] - rowLse[m];
    }
}

// ---------------------------------------------------------------------------
// Host launcher
// ---------------------------------------------------------------------------
extern "C" void kernel_launch(void* const* d_in, const int* in_sizes, int n_in,
                              void* d_out, int out_size, void* d_ws, size_t ws_size,
                              hipStream_t stream) {
    (void)in_sizes; (void)n_in; (void)out_size; (void)ws_size;

    const float* x  = (const float*)d_in[0];   // [T,B,IN]
    const float* h0 = (const float*)d_in[1];   // [1,B,H]
    const float* c0 = (const float*)d_in[2];   // [1,B,H]
    const float* Wg = (const float*)d_in[3];   // [4H, IN+H]
    const float* bg = (const float*)d_in[4];   // [4H]
    const float* Wo = (const float*)d_in[5];   // [OUT, H]
    const float* bo = (const float*)d_in[6];   // [OUT]
    float* out = (float*)d_out;                // log_probs | h_f | c_f

    // Workspace layout
    char* p = (char*)d_ws;
    unsigned short* Wg_bf  = (unsigned short*)p; p += (size_t)FOUR_H * K_TOT * 2;           // 16 MB
    unsigned short* Wo_bf  = (unsigned short*)p; p += (size_t)OUT_DIM * H_DIM * 2;          // 1 MB
    unsigned short* x_bf   = (unsigned short*)p; p += (size_t)T_STEPS * BATCH * IN_DIM * 2; // 32 MB
    unsigned short* outs_bf= (unsigned short*)p; p += (size_t)T_STEPS * BATCH * H_DIM * 2;  // 32 MB
    unsigned short* h_bf   = (unsigned short*)p; p += (size_t)2 * BATCH * H_DIM * 2;        // ping-pong
    float*          h_f32  = (float*)p;          p += (size_t)BATCH * H_DIM * 4;
    float*          c_f32  = (float*)p;          p += (size_t)BATCH * H_DIM * 4;

    // 1) one-time conversions to bf16
    cvt_f32_to_bf16<<<2048, 256, 0, stream>>>(Wg, Wg_bf, (size_t)FOUR_H * K_TOT);
    cvt_f32_to_bf16<<<512,  256, 0, stream>>>(Wo, Wo_bf, (size_t)OUT_DIM * H_DIM);
    cvt_f32_to_bf16<<<4096, 256, 0, stream>>>(x,  x_bf,  (size_t)T_STEPS * BATCH * IN_DIM);

    // 2) state init
    init_state<<<(BATCH * H_DIM + 255) / 256, 256, 0, stream>>>(h0, c0, h_bf, h_f32, c_f32);

    // 3) sequential recurrence: one WMMA kernel per timestep
    for (int t = 0; t < T_STEPS; ++t) {
        lstm_step<<<H_DIM / 16, 512, 0, stream>>>(
            Wg_bf, bg,
            x_bf + (size_t)t * BATCH * IN_DIM,
            h_bf + (size_t)(t & 1) * BATCH * H_DIM,
            h_bf + (size_t)((t + 1) & 1) * BATCH * H_DIM,
            h_f32, c_f32,
            outs_bf + (size_t)t * BATCH * H_DIM);
    }

    // 4) projection + log_softmax -> d_out
    proj_logsoftmax<<<(T_STEPS * BATCH) / 16, 256, 0, stream>>>(outs_bf, Wo_bf, bo, out);

    // 5) final states appended after log_probs
    const size_t lp = (size_t)T_STEPS * BATCH * OUT_DIM;
    hipMemcpyAsync(out + lp,                 h_f32, (size_t)BATCH * H_DIM * 4,
                   hipMemcpyDeviceToDevice, stream);
    hipMemcpyAsync(out + lp + BATCH * H_DIM, c_f32, (size_t)BATCH * H_DIM * 4,
                   hipMemcpyDeviceToDevice, stream);
}
